// TemporalGraphConv_23158463660306
// MI455X (gfx1250) — matrix operand order repaired and checked by reading, not compiled, not run on hardware
//
#include <hip/hip_runtime.h>

typedef __attribute__((ext_vector_type(16))) _Float16 v16h;
typedef __attribute__((ext_vector_type(8)))  float    v8f;

#define BB 4
#define NN 4096
#define QQ 4096

// ---------------------------------------------------------------------------
// Brute-force kNN: one thread per query, sorted insertion of top-K (registers)
// ---------------------------------------------------------------------------
template <int D, int KK>
__global__ __launch_bounds__(256)
void knn_kernel(const float* __restrict__ qp, const float* __restrict__ kp,
                int* __restrict__ outIdx, int Qn, int Nn) {
  int t = blockIdx.x * blockDim.x + threadIdx.x;
  int b = t / Qn, q = t % Qn;
  float qv[D];
#pragma unroll
  for (int d = 0; d < D; ++d) qv[d] = qp[((size_t)b * Qn + q) * D + d];
  float bd[KK]; int bi[KK];
#pragma unroll
  for (int j = 0; j < KK; ++j) { bd[j] = 3.4e38f; bi[j] = 0; }
  const float* kb = kp + (size_t)b * Nn * D;
  for (int n = 0; n < Nn; ++n) {
    float dist = 0.f;
#pragma unroll
    for (int d = 0; d < D; ++d) { float df = qv[d] - kb[(size_t)n * D + d]; dist += df * df; }
    if (dist < bd[KK - 1]) {
      bd[KK - 1] = dist; bi[KK - 1] = n;
#pragma unroll
      for (int j = KK - 1; j > 0; --j) {
        if (bd[j] < bd[j - 1]) {
          float td = bd[j]; bd[j] = bd[j - 1]; bd[j - 1] = td;
          int   ti = bi[j]; bi[j] = bi[j - 1]; bi[j - 1] = ti;
        }
      }
    }
  }
#pragma unroll
  for (int j = 0; j < KK; ++j) outIdx[((size_t)b * Qn + q) * KK + j] = bi[j];
}

// ---------------------------------------------------------------------------
// Weight pre-conversion: W[c][o] f32  ->  Wh[o][KPAD] f16 (transposed, 0-padded)
// so B fragments become contiguous 32B-aligned v16h loads.
// ---------------------------------------------------------------------------
__global__ __launch_bounds__(256)
void wcvt_kernel(const float* __restrict__ W, _Float16* __restrict__ out,
                 int CIN, int OUT, int KPAD) {
  int t = blockIdx.x * blockDim.x + threadIdx.x;
  if (t >= OUT * KPAD) return;
  int o = t / KPAD, c = t % KPAD;
  out[(size_t)o * KPAD + c] = (c < CIN) ? (_Float16)W[(size_t)c * OUT + o] : (_Float16)0.f;
}

// ---------------------------------------------------------------------------
// Fused gather + [concat f0|f1|f2|rel] x W + bias + ReLU + max-over-K via WMMA.
// Rows of the GEMM = (query, neighbor) pairs. 128 threads = 4 waves; each wave
// owns one 16-row M-tile, reuses its A fragments across OUT/16 N-tiles.
// K_==1 degenerates to a dense per-node GEMM (combine MLP).
// ---------------------------------------------------------------------------
template <int K_, int C0, int C1, int C2, int R, int OUT, bool DORELU>
__global__ __launch_bounds__(128)
void fused_wmma_kernel(const float* __restrict__ f0, const float* __restrict__ f1,
                       const float* __restrict__ f2,
                       const float* __restrict__ qpos, const float* __restrict__ kpos,
                       const int* __restrict__ idx,
                       const _Float16* __restrict__ Wh, const float* __restrict__ bias,
                       float* __restrict__ out, int Qn, int Nn) {
  constexpr int CIN    = C0 + C1 + C2 + R;
  constexpr int KPAD   = (CIN + 31) & ~31;
  constexpr int NCHUNK = KPAD / 32;

  const int lane = threadIdx.x & 31;
  const int wave = threadIdx.x >> 5;
  const int tile = blockIdx.x * 4 + wave;
  const int tpb  = (Qn * K_) >> 4;          // 16-row tiles per batch
  const int b    = tile / tpb;
  const int rowBase = (tile % tpb) * 16;

  // Per-lane A-row: lanes L and L+16 share row M = L&15 (different K halves).
  const int M = lane & 15;
  const int g = rowBase + M;
  const int q = g / K_;
  const int k = g % K_;
  int n;
  if (K_ == 1) n = q;
  else         n = idx[((size_t)b * Qn + q) * K_ + k];

  auto fetch = [&](int c) -> float {        // concat(f0, f1, f2, kpos-qpos), 0-pad
    if (c < C0) return f0[((size_t)b * Nn + n) * C0 + c];
    if (C1 > 0 && c < C0 + C1) return f1[((size_t)b * Nn + n) * C1 + (c - C0)];
    if (C2 > 0 && c < C0 + C1 + C2) return f2[((size_t)b * Nn + n) * C2 + (c - C0 - C1)];
    if (R > 0 && c < CIN) {
      int d = c - C0 - C1 - C2;
      return kpos[((size_t)b * Nn + n) * R + d] - qpos[((size_t)b * Qn + q) * R + d];
    }
    return 0.f;
  };

  // Build all A fragments once (contiguous k = 16*half + e, matching B).
  const int abase = (lane < 16) ? 0 : 16;
  v16h afrag[NCHUNK];
#pragma unroll
  for (int kc = 0; kc < NCHUNK; ++kc) {
#pragma unroll
    for (int e = 0; e < 16; ++e) {
      int c = kc * 32 + abase + e;
      afrag[kc][e] = (_Float16)fetch(c);
    }
  }

  const int bbase = (lane < 16) ? 0 : 16;   // contiguous k = 16*half + e, matching A
  const int col   = lane & 15;
#pragma unroll
  for (int nt = 0; nt < OUT / 16; ++nt) {
    const int ocol = nt * 16 + col;
    const _Float16* brow = Wh + (size_t)ocol * KPAD + bbase;
    v8f acc = {0.f, 0.f, 0.f, 0.f, 0.f, 0.f, 0.f, 0.f};
#pragma unroll
    for (int kc = 0; kc < NCHUNK; ++kc) {
      v16h bfrag = *reinterpret_cast<const v16h*>(brow + kc * 32);
      acc = __builtin_amdgcn_wmma_f32_16x16x32_f16(false, afrag[kc], false, bfrag,
                                                   (short)0, acc, false, false);
    }
    const float bv = bias[ocol];
    if (K_ == 1) {
#pragma unroll
      for (int r = 0; r < 8; ++r) {
        float v = acc[r] + bv;
        if (DORELU) v = fmaxf(v, 0.f);
        int q2 = rowBase + r + ((lane < 16) ? 0 : 8);
        out[((size_t)b * Qn + q2) * OUT + ocol] = v;
      }
    } else {
      float m = -3.4e38f;
#pragma unroll
      for (int r = 0; r < 8; ++r) m = fmaxf(m, fmaxf(acc[r] + bv, 0.f));
      if (K_ == 16) {                       // one query per tile: fold lane halves
        m = fmaxf(m, __shfl_xor(m, 16, 32));
        if (lane < 16) {
          int q2 = rowBase >> 4;
          out[((size_t)b * Qn + q2) * OUT + ocol] = m;
        }
      } else {                              // K_==8: two queries per tile, by half
        int q2 = (rowBase >> 3) + ((lane < 16) ? 0 : 1);
        out[((size_t)b * Qn + q2) * OUT + ocol] = m;
      }
    }
  }
}

// ---------------------------------------------------------------------------
extern "C" void kernel_launch(void* const* d_in, const int* in_sizes, int n_in,
                              void* d_out, int out_size, void* d_ws, size_t ws_size,
                              hipStream_t stream) {
  (void)in_sizes; (void)n_in; (void)out_size; (void)ws_size;
  // setup_inputs() order: data, ids, space_pts, time_pts, query_pts, params.
  // params flattened jax-style (sorted keys): Wq, bq, then layers[0..2] each as
  // (Wc1, Wc2, Ws, Wt, bc1, bc2, bs, bt).
  const float* data  = (const float*)d_in[0];
  const float* space = (const float*)d_in[2];
  const float* timep = (const float*)d_in[3];
  const float* query = (const float*)d_in[4];
  const float* Wq    = (const float*)d_in[5];
  const float* bq    = (const float*)d_in[6];

  char* ws = (char*)d_ws;
  size_t off = 0;
  auto alloc = [&](size_t bytes) -> void* {
    void* p = ws + off; off += (bytes + 255) & ~(size_t)255; return p;
  };
  int*   sidx = (int*)alloc(sizeof(int) * BB * NN * 16);
  int*   tidx = (int*)alloc(sizeof(int) * BB * NN * 8);
  int*   qidx = (int*)alloc(sizeof(int) * BB * QQ * 16);
  float* sn   = (float*)alloc(sizeof(float) * BB * NN * 64);
  float* tn   = (float*)alloc(sizeof(float) * BB * NN * 64);
  float* hb   = (float*)alloc(sizeof(float) * BB * NN * 128);
  float* xA   = (float*)alloc(sizeof(float) * BB * NN * 64);
  float* xB   = (float*)alloc(sizeof(float) * BB * NN * 64);
  // f16 transposed+padded weights: [OUT][KPAD]
  _Float16* Wsh[3]; _Float16* Wth[3]; _Float16* Wc1h[3]; _Float16* Wc2h[3];
  for (int l = 0; l < 3; ++l) {
    Wsh[l]  = (_Float16*)alloc(sizeof(_Float16) * 64 * 96);    // CIN 67  -> 96
    Wth[l]  = (_Float16*)alloc(sizeof(_Float16) * 64 * 160);   // CIN 136 -> 160
    Wc1h[l] = (_Float16*)alloc(sizeof(_Float16) * 128 * 192);  // CIN 192 -> 192
    Wc2h[l] = (_Float16*)alloc(sizeof(_Float16) * 64 * 128);   // CIN 128 -> 128
  }
  _Float16* Wqh = (_Float16*)alloc(sizeof(_Float16) * 64 * 96);

  auto cvt = [&](const float* W, _Float16* Wh, int cin, int outd) {
    int kpad = (cin + 31) & ~31;
    int total = outd * kpad;
    wcvt_kernel<<<(total + 255) / 256, 256, 0, stream>>>(W, Wh, cin, outd, kpad);
  };
  for (int l = 0; l < 3; ++l) {
    cvt((const float*)d_in[7 + 8 * l + 2], Wsh[l],  67,  64);
    cvt((const float*)d_in[7 + 8 * l + 3], Wth[l],  136, 64);
    cvt((const float*)d_in[7 + 8 * l + 0], Wc1h[l], 192, 128);
    cvt((const float*)d_in[7 + 8 * l + 1], Wc2h[l], 128, 64);
  }
  cvt(Wq, Wqh, 67, 64);

  knn_kernel<3, 16><<<(BB * NN) / 256, 256, 0, stream>>>(space, space, sidx, NN, NN);
  knn_kernel<8, 8 ><<<(BB * NN) / 256, 256, 0, stream>>>(timep, timep, tidx, NN, NN);
  knn_kernel<3, 16><<<(BB * QQ) / 256, 256, 0, stream>>>(query, space, qidx, QQ, NN);

  const float* xin = data;
  for (int l = 0; l < 3; ++l) {
    const float* bc1 = (const float*)d_in[7 + 8 * l + 4];
    const float* bc2 = (const float*)d_in[7 + 8 * l + 5];
    const float* bs  = (const float*)d_in[7 + 8 * l + 6];
    const float* bt  = (const float*)d_in[7 + 8 * l + 7];

    // Spatial GraphConv: [N*16, 64+3] x [67,64] -> relu -> max_k
    fused_wmma_kernel<16, 64, 0, 0, 3, 64, true>
        <<<(BB * NN * 16 / 16) / 4, 128, 0, stream>>>(
            xin, nullptr, nullptr, space, space, sidx, Wsh[l], bs, sn, NN, NN);
    // Temporal GraphConv on concat(x, sn): [N*8, 128+8] x [136,64]
    fused_wmma_kernel<8, 64, 64, 0, 8, 64, true>
        <<<(BB * NN * 8 / 16) / 4, 128, 0, stream>>>(
            xin, sn, nullptr, timep, timep, tidx, Wth[l], bt, tn, NN, NN);
    // Combine MLP layer 1: concat(x, sn, tn) [N,192] x [192,128] + relu
    fused_wmma_kernel<1, 64, 64, 64, 0, 128, true>
        <<<(BB * NN / 16) / 4, 128, 0, stream>>>(
            xin, sn, tn, nullptr, nullptr, nullptr, Wc1h[l], bc1, hb, NN, NN);
    // Combine MLP layer 2: [N,128] x [128,64] (no relu)
    float* xnext = (l == 1) ? xB : xA;
    fused_wmma_kernel<1, 128, 0, 0, 0, 64, false>
        <<<(BB * NN / 16) / 4, 128, 0, stream>>>(
            hb, nullptr, nullptr, nullptr, nullptr, nullptr, Wc2h[l], bc2, xnext, NN, NN);
    xin = xnext;
  }

  // Target conv over query points: gather x via qidx, rel = space - query
  fused_wmma_kernel<16, 64, 0, 0, 3, 64, true>
      <<<(BB * QQ * 16 / 16) / 4, 128, 0, stream>>>(
          xin, nullptr, nullptr, query, space, qidx, Wqh, bq, (float*)d_out, QQ, NN);
}
